// TransFM_790273982845
// MI455X (gfx1250) — matrix operand order repaired and checked by reading, not compiled, not run on hardware
//
#include <hip/hip_runtime.h>
#include <hip/hip_bf16.h>

typedef __attribute__((ext_vector_type(16))) __bf16         v16bf;
typedef __attribute__((ext_vector_type(8)))  float          v8f;
typedef __attribute__((ext_vector_type(16))) unsigned short v16u;
typedef __attribute__((ext_vector_type(4)))  unsigned int   u32x4;
typedef __attribute__((ext_vector_type(8)))  int            i32x8;
typedef __attribute__((ext_vector_type(4)))  int            i32x4;

#define B_   2048
#define UD   10000
#define XD   20000
#define K_   50000
#define NF   128
#define NE   272          // 128 (V) + 128 (T) + vv,tt,vt,W,ones + pad -> 17 tiles of 16
#define NT   17
#define KT   1563         // ceil(50000/32); K padded to 50016
#define SPLITS 16
#define WAVES 8           // waves per GEMM workgroup (share B via LDS)

// Per K-tile packed-B footprint: 17 tiles * 512 halves = 8704 halves = 17408 bytes (per plane)
#define KTILE_HALVES 8704
#define KTILE_BYTES  17408
#define STAGE_HALVES (2 * KTILE_HALVES)   // hi plane + lo plane
#define STAGE_BYTES  (2 * KTILE_BYTES)

constexpr size_t alignup(size_t x) { return (x + 255) & ~size_t(255); }
constexpr size_t OFF_VV  = 0;
constexpr size_t OFF_TT  = alignup(OFF_VV + (size_t)K_ * 4);
constexpr size_t OFF_VT  = alignup(OFF_TT + (size_t)K_ * 4);
constexpr size_t OFF_BHI = alignup(OFF_VT + (size_t)K_ * 4);
constexpr size_t BH_BYTES = (size_t)KT * KTILE_BYTES;
constexpr size_t OFF_BLO = alignup(OFF_BHI + BH_BYTES);
constexpr size_t OFF_OUT = alignup(OFF_BLO + BH_BYTES);
constexpr size_t OUT_BYTES = (size_t)2 * B_ * NE * 4;

#if __has_builtin(__builtin_amdgcn_tensor_load_to_lds) && __has_builtin(__builtin_amdgcn_s_wait_tensorcnt)
#define HAVE_TDM 1
#else
#define HAVE_TDM 0
#endif

// ---- fp32 <-> bf16 (round-to-nearest-even) helpers ----
__device__ __forceinline__ unsigned short f2bf(float f) {
    unsigned u = __float_as_uint(f);
    unsigned r = (u + 0x7fffu + ((u >> 16) & 1u)) >> 16;
    return (unsigned short)r;
}
__device__ __forceinline__ float bf2f(unsigned short h) {
    return __uint_as_float(((unsigned)h) << 16);
}

// Load 8 contiguous fp32 "z" elements for row m starting at virtual column k8.
// k8 is always a multiple of 8; segment boundaries (10000, 30000) are multiples
// of 8 so each run lies entirely within one source array.
__device__ __forceinline__ void loadrun8(const float* __restrict__ u,
                                         const float* __restrict__ xp,
                                         const float* __restrict__ xx,
                                         int m, int k8, float* f) {
    if (k8 >= K_) {
        #pragma unroll
        for (int i = 0; i < 8; ++i) f[i] = 0.f;
        return;
    }
    const float* p;
    if      (k8 >= UD + XD) p = xx + (size_t)m * XD + (k8 - (UD + XD));
    else if (k8 >= UD)      p = xp + (size_t)m * XD + (k8 - UD);
    else                    p = u  + (size_t)m * UD + k8;
    float4 a = *(const float4*)p;
    float4 b = *(const float4*)(p + 4);
    f[0]=a.x; f[1]=a.y; f[2]=a.z; f[3]=a.w;
    f[4]=b.x; f[5]=b.y; f[6]=b.z; f[7]=b.w;
}

// ---- 1) row-wise Gram diagonals: vv = sum(V*V,1), tt, vt ----
__global__ __launch_bounds__(256)
void gram_kernel(const float* __restrict__ V, const float* __restrict__ T,
                 float* __restrict__ vv, float* __restrict__ tt, float* __restrict__ vt) {
    int wave = threadIdx.x >> 5, lane = threadIdx.x & 31;
    int k = blockIdx.x * 8 + wave;
    if (k >= K_) return;
    const float* vr = V + (size_t)k * NF;
    const float* tr = T + (size_t)k * NF;
    float sv = 0.f, st = 0.f, sx = 0.f;
    #pragma unroll
    for (int c = lane; c < NF; c += 32) {
        float v = vr[c], t = tr[c];
        sv += v * v; st += t * t; sx += v * t;
    }
    #pragma unroll
    for (int o = 16; o > 0; o >>= 1) {
        sv += __shfl_xor(sv, o, 32);
        st += __shfl_xor(st, o, 32);
        sx += __shfl_xor(sx, o, 32);
    }
    if (lane == 0) { vv[k] = sv; tt[k] = st; vt[k] = sx; }
}

// ---- 2) pack extended B-matrix [K, 272] into WMMA B (32x16 bf16) lane layout ----
// Tile (kt, nt) -> 512 halves; lane holds 16 contiguous halves:
//   n = nt*16 + (lane&15);  k = kt*32 + (lane>>4)*16 + e,  e = 0..15
__global__ __launch_bounds__(256)
void pack_kernel(const float* __restrict__ V, const float* __restrict__ T,
                 const float* __restrict__ vv, const float* __restrict__ tt,
                 const float* __restrict__ vt, const float* __restrict__ W,
                 unsigned short* __restrict__ Bhi, unsigned short* __restrict__ Blo) {
    int wave = threadIdx.x >> 5, lane = threadIdx.x & 31;
    int tile = blockIdx.x * 8 + wave;
    if (tile >= KT * NT) return;
    int kt = tile / NT, nt = tile % NT;
    int khalf = lane >> 4, ncol = nt * 16 + (lane & 15);
    unsigned short* ph = Bhi + (size_t)tile * 512 + lane * 16;
    unsigned short* pl = Blo + (size_t)tile * 512 + lane * 16;
    #pragma unroll
    for (int e = 0; e < 16; ++e) {
        int k = kt * 32 + khalf * 16 + e;
        float val = 0.f;
        if (k < K_) {
            if      (ncol < 128)  val = V[(size_t)k * NF + ncol];
            else if (ncol < 256)  val = T[(size_t)k * NF + (ncol - 128)];
            else if (ncol == 256) val = vv[k];
            else if (ncol == 257) val = tt[k];
            else if (ncol == 258) val = vt[k];
            else if (ncol == 259) val = W[k];
            else if (ncol == 260) val = 1.0f;
        }
        unsigned short h = f2bf(val);
        ph[e] = h;
        pl[e] = f2bf(val - bf2f(h));
    }
}

#if HAVE_TDM
// Issue one TDM load: a 2-row tile (hi plane row, lo plane row) of one K-tile of
// packed B into LDS at lds_off_bytes. D# built per CDNA5 ISA 8.3-8.6.
// This toolchain's builtin takes 6 args: (g0, g1, g2, g3, extra_i32x8, cpol).
__device__ __forceinline__ void tdm_stage(const unsigned short* Bhi,
                                          const unsigned short* Blo,
                                          unsigned lds_off_bytes, int kt) {
    unsigned long long ga = (unsigned long long)(uintptr_t)Bhi
                          + (unsigned long long)kt * (unsigned long long)KTILE_BYTES;
    // Row stride (in 8-byte elements) between hi plane and lo plane in global mem.
    unsigned long long strideEl =
        ((unsigned long long)((const char*)Blo - (const char*)Bhi)) >> 3;

    // Group 0: count=1 | lds_addr | 57-bit global_addr | type=2
    u32x4 g0 = { 1u,
                 lds_off_bytes,
                 (unsigned)(ga & 0xffffffffu),
                 (unsigned)((ga >> 32) & 0x01ffffffu) | (2u << 30) };
    // Group 1: data_size=8B; tensor_dim0=2176, tensor_dim1=2;
    //          tile_dim0=2176, tile_dim1=2, tile_dim2=0; stride0=strideEl
    i32x8 g1 = { (int)(3u << 16),                    // wg_mask=0, data_size=3 (8B), no flags
                 (int)((2176u & 0xffffu) << 16),     // [15:0] atomic addr=0, [31:16] dim0 lo
                 (int)(2u << 16),                    // [15:0] dim0 hi=0, [31:16] dim1 lo=2
                 (int)(2176u << 16),                 // [15:0] dim1 hi=0, [31:16] tile_dim0
                 (int)2,                             // [15:0] tile_dim1=2, [31:16] tile_dim2=0
                 (int)(unsigned)(strideEl & 0xffffffffu),   // stride0 lo32
                 (int)((strideEl >> 32) & 0xffffu),         // stride0 hi16 | stride1 lo16=0
                 0 };                                       // stride1 hi32=0
    // Groups 2/3: higher dims = 1, higher tiles = 0 (unused)
    i32x4 g2 = { 1, 1, 0, 0 };               // tensor_dim2=1, tensor_dim3=1
    i32x4 g3 = { 0, (int)(1u << 16), 0, 0 }; // stride3=0, tensor_dim4=1, tile_dim4=0
    i32x8 gx = { 0, 0, 0, 0, 0, 0, 0, 0 };   // extra operand (zero-filled)
    __builtin_amdgcn_tensor_load_to_lds(g0, g1, g2, g3, gx, 0);
}
#else
// Fallback: cooperative global->reg->LDS copy of one K-tile (hi+lo planes).
__device__ __forceinline__ void copy_stage(const unsigned short* __restrict__ Bhi,
                                           const unsigned short* __restrict__ Blo,
                                           unsigned short* lbuf, int kt, int tid) {
    const unsigned short* sh = Bhi + (size_t)kt * KTILE_HALVES;
    const unsigned short* sl = Blo + (size_t)kt * KTILE_HALVES;
    for (int idx = tid; idx < 2 * 1088; idx += 256) {
        if (idx < 1088)
            *(uint4*)(lbuf + idx * 8) = *(const uint4*)(sh + idx * 8);
        else
            *(uint4*)(lbuf + KTILE_HALVES + (idx - 1088) * 8) =
                *(const uint4*)(sl + (idx - 1088) * 8);
    }
}
#endif

// ---- 3) fused GEMM: out[pol][B, 272] += z_pol @ Bext ----
// 8 waves per block share TDM-staged B in LDS (double buffered); each wave owns
// one 16-row M-tile; split-bf16 (hi/lo) 3-product scheme; split-K via atomics.
__global__ __launch_bounds__(256)
void gemm_kernel(const float* __restrict__ u, const float* __restrict__ xp,
                 const float* __restrict__ xpos, const float* __restrict__ xneg,
                 const unsigned short* __restrict__ Bhi,
                 const unsigned short* __restrict__ Blo,
                 float* __restrict__ out) {
    __shared__ unsigned short ldsb[2 * STAGE_HALVES];   // 2 x 34816 B = 69632 B

    int tid  = threadIdx.x;
    int wave = tid >> 5, lane = tid & 31;
    int bid  = blockIdx.x;
    int s    = bid & (SPLITS - 1);
    int mg   = (bid >> 4) & 15;
    int pol  = bid >> 8;
    const float* xx = pol ? xneg : xpos;

    int mt   = mg * WAVES + wave;
    int half = lane >> 4;             // A/B lane group (K 0..15 vs 16..31)
    int m    = mt * 16 + (lane & 15);

    v8f zero = {};
    v8f acc[NT];
    #pragma unroll
    for (int i = 0; i < NT; ++i) acc[i] = zero;

    constexpr int CH = (KT + SPLITS - 1) / SPLITS;  // 98
    int kt0 = s * CH;
    int kt1 = kt0 + CH; if (kt1 > KT) kt1 = KT;

#if HAVE_TDM
    unsigned lds_base = (unsigned)(uintptr_t)(void*)ldsb;  // generic->LDS offset (low 32b)
    if (wave == 0) tdm_stage(Bhi, Blo, lds_base, kt0);
#else
    copy_stage(Bhi, Blo, ldsb, kt0, tid);
#endif

    for (int kt = kt0; kt < kt1; ++kt) {
        int cur = (kt - kt0) & 1;
#if HAVE_TDM
        __builtin_amdgcn_s_wait_tensorcnt(0);
#endif
        __syncthreads();                       // staged B for `cur` visible to all waves
        if (kt + 1 < kt1) {
#if HAVE_TDM
            if (wave == 0)
                tdm_stage(Bhi, Blo, lds_base + (unsigned)((cur ^ 1) * STAGE_BYTES), kt + 1);
#else
            copy_stage(Bhi, Blo, ldsb + (cur ^ 1) * STAGE_HALVES, kt + 1, tid);
#endif
        }

        // A tile (16x32 bf16 layout): lane e-list = {half*16+0..15}
        float f[16];
        int kb = kt * 32;
        loadrun8(u, xp, xx, m, kb + half * 16, f);
        loadrun8(u, xp, xx, m, kb + half * 16 + 8, f + 8);
        v16u hiu, lou;
        #pragma unroll
        for (int e = 0; e < 16; ++e) {
            unsigned short h = f2bf(f[e]);
            hiu[e] = h;
            lou[e] = f2bf(f[e] - bf2f(h));
        }
        v16bf ahi = __builtin_bit_cast(v16bf, hiu);
        v16bf alo = __builtin_bit_cast(v16bf, lou);

        const unsigned short* lb = ldsb + cur * STAGE_HALVES + lane * 16;
        #pragma unroll
        for (int nt = 0; nt < NT; ++nt) {
            v16bf bhi = __builtin_bit_cast(v16bf, *(const v16u*)(lb + nt * 512));
            v16bf blo = __builtin_bit_cast(v16bf, *(const v16u*)(lb + KTILE_HALVES + nt * 512));
            v8f a = acc[nt];
            a = __builtin_amdgcn_wmma_f32_16x16x32_bf16(false, ahi, false, bhi, (short)0, a, false, false);
            a = __builtin_amdgcn_wmma_f32_16x16x32_bf16(false, ahi, false, blo, (short)0, a, false, false);
            a = __builtin_amdgcn_wmma_f32_16x16x32_bf16(false, alo, false, bhi, (short)0, a, false, false);
            acc[nt] = a;
        }
        __syncthreads();                       // all waves done with `cur` before reuse
    }

    // D layout: VGPR r, lane l -> M = r + 8*(l>=16), N = l&15
    float* orow = out + (size_t)pol * B_ * NE;
    int n0 = lane & 15;
    #pragma unroll
    for (int nt = 0; nt < NT; ++nt) {
        #pragma unroll
        for (int r = 0; r < 8; ++r) {
            atomicAdd(&orow[(size_t)(mt * 16 + r + half * 8) * NE + nt * 16 + n0],
                      acc[nt][r]);
        }
    }
}

// ---- 4) epilogue ----
__global__ __launch_bounds__(256)
void finalize_kernel(const float* __restrict__ out, const float* __restrict__ bptr,
                     float* __restrict__ pred) {
    int wave = threadIdx.x >> 5, lane = threadIdx.x & 31;
    int b = blockIdx.x * 8 + wave;
    if (b >= B_) return;
    const float* rp = out + (size_t)b * NE;                     // pos row
    const float* rn = out + (size_t)B_ * NE + (size_t)b * NE;   // neg row
    float p5 = 0, p6 = 0, pd = 0, n5 = 0, n6 = 0, nd = 0;
    #pragma unroll
    for (int c = lane; c < NF; c += 32) {
        float vx = rp[c], tx = rp[128 + c];
        p5 += vx * vx; p6 += tx * vx; pd += tx * tx;
        float vy = rn[c], ty = rn[128 + c];
        n5 += vy * vy; n6 += ty * vy; nd += ty * ty;
    }
    #pragma unroll
    for (int o = 16; o > 0; o >>= 1) {
        p5 += __shfl_xor(p5, o, 32); p6 += __shfl_xor(p6, o, 32); pd += __shfl_xor(pd, o, 32);
        n5 += __shfl_xor(n5, o, 32); n6 += __shfl_xor(n6, o, 32); nd += __shfl_xor(nd, o, 32);
    }
    if (lane == 0) {
        float bias = bptr[0];
        float lin_neg = rn[259] + bias;          // reference reuses NEG linear term
        float zsp = rp[260], zsn = rn[260];
        float qp = 0.5f * (zsp * rp[256] + 2.f * zsp * rp[257] + 2.f * zsp * rp[258]
                           - 2.f * p5 - 2.f * p6) - 0.5f * pd;
        float qn = 0.5f * (zsn * rn[256] + 2.f * zsn * rn[257] + 2.f * zsn * rn[258]
                           - 2.f * n5 - 2.f * n6) - 0.5f * nd;
        pred[b]      = lin_neg + qp;
        pred[B_ + b] = lin_neg + qn;
    }
}

extern "C" void kernel_launch(void* const* d_in, const int* in_sizes, int n_in,
                              void* d_out, int out_size, void* d_ws, size_t ws_size,
                              hipStream_t stream) {
    (void)in_sizes; (void)n_in; (void)out_size; (void)ws_size;
    const float* u    = (const float*)d_in[0];
    const float* xp   = (const float*)d_in[1];
    const float* xpos = (const float*)d_in[2];
    const float* xneg = (const float*)d_in[3];
    const float* V    = (const float*)d_in[4];
    const float* T    = (const float*)d_in[5];
    const float* W    = (const float*)d_in[6];
    const float* bias = (const float*)d_in[7];
    float* pred = (float*)d_out;

    char* ws = (char*)d_ws;
    float* vv = (float*)(ws + OFF_VV);
    float* tt = (float*)(ws + OFF_TT);
    float* vt = (float*)(ws + OFF_VT);
    unsigned short* Bhi = (unsigned short*)(ws + OFF_BHI);
    unsigned short* Blo = (unsigned short*)(ws + OFF_BLO);
    float* outbuf = (float*)(ws + OFF_OUT);

    (void)hipMemsetAsync(outbuf, 0, OUT_BYTES, stream);
    gram_kernel<<<(K_ + 7) / 8, 256, 0, stream>>>(V, T, vv, tt, vt);
    pack_kernel<<<(KT * NT + 7) / 8, 256, 0, stream>>>(V, T, vv, tt, vt, W, Bhi, Blo);
    gemm_kernel<<<2 * 16 * SPLITS, 256, 0, stream>>>(u, xp, xpos, xneg, Bhi, Blo, outbuf);
    finalize_kernel<<<(B_ + 7) / 8, 256, 0, stream>>>(outbuf, bias, pred);
}
